// ScaledDotProductAttention_encoder_88656714924832
// MI455X (gfx1250) — matrix operand order repaired and checked by reading, not compiled, not run on hardware
//
#include <hip/hip_runtime.h>
#include <hip/hip_bf16.h>

// ScaledDotProductAttention (B=8,H=16,S=1024,d=16) for gfx1250.
// Flash-attention single pass; QK^T computed TRANSPOSED (K @ Q^T) so each lane
// owns a full query row of scores:
//   - softmax reductions are in-lane (+1 cross-half shfl) instead of 64 bpermutes
//   - P maps directly onto the PV WMMA A-operand: zero data movement, no LDS
//   - bias/mask loads are contiguous per lane (b128 / b64)
// v_wmma_f32_16x16x32_f16 for both GEMMs; f32 softmax; V staged in LDS as f16^T.

typedef __attribute__((ext_vector_type(16))) _Float16 v16h;
typedef __attribute__((ext_vector_type(8)))  float    v8f;

#define Bn 8
#define Hn 16
#define Sn 1024
#define Dk 16

#define WAVES 8            // waves per workgroup (wave32)
#define QT    (WAVES * 16) // queries per workgroup = 128
#define VROW  1032         // padded Vt row (halves): 516 dwords/row -> 16 distinct banks

__global__ __launch_bounds__(WAVES * 32)
void sdpa_flash_kernel(const float* __restrict__ Q,
                       const float* __restrict__ K,
                       const float* __restrict__ V,
                       const unsigned char* __restrict__ mask, // bool [B,1,S,S]
                       const float* __restrict__ bias,         // [B,1,S,S]
                       float* __restrict__ out)                // [B,H,S,d]
{
    __shared__ _Float16 Vt[16 * VROW]; // V transposed, f16: 33 KB

    const int tid  = threadIdx.x;
    const int wave = tid >> 5;
    const int lane = tid & 31;
    const int row  = lane & 15; // position within 16-lane half
    const int half = lane >> 4; // 0 or 1

    const int nQB  = Sn / QT;          // 8 q-blocks per (b,h)
    const int bh   = blockIdx.x / nQB; // b*H + h
    const int qblk = blockIdx.x % nQB;
    const int b    = bh / Hn;

    const float* Qp = Q + (size_t)bh * Sn * Dk;
    const float* Kp = K + (size_t)bh * Sn * Dk;
    const float* Vp = V + (size_t)bh * Sn * Dk;
    float*       Op = out + (size_t)bh * Sn * Dk;

    // ---- stage V as f16, transposed: Vt[d][key] (cooperative) ----
    for (int key = tid; key < Sn; key += WAVES * 32) {
        const float* vp = Vp + key * Dk;
#pragma unroll
        for (int d = 0; d < 16; ++d)
            Vt[d * VROW + key] = (_Float16)vp[d];
    }
    __syncthreads();

    const int q0   = qblk * QT + wave * 16; // this wave's first query row
    const int qrow = q0 + row;              // this LANE's query row (scores^T)

    // ---- Q as B-operand (32x16): lane holds column N=row -> Q[qrow][0:16] ----
    // (elements 8..15 of every lane multiply against A's zeroed kred 16..31 -> finite garbage is fine)
    v16h b_q;
    {
        const float* qp = Qp + (size_t)qrow * Dk;
#pragma unroll
        for (int i = 0; i < 16; ++i)
            b_q[i] = (_Float16)(qp[(i + half * 8) & 15] * 0.25f); // 1/sqrt(16) folded in
    }

    // Per-lane bases: all inner-loop loads are base + immediate offsets.
    const float*         KpL = Kp + row * Dk + half * 8; // A-op: key=kbt+row, d=8h..8h+7
    const size_t         qso = (size_t)b * Sn * Sn + (size_t)qrow * Sn + half * 8;
    const float*         BpL = bias + qso;
    const unsigned char* MpL = mask + qso;

    v8f   acc = {};                 // context accumulator (D: q=q0+r+8h, d=row)
    float m = -INFINITY, l = 0.0f;  // per-lane-row softmax stats

    for (int kb = 0; kb < Sn; kb += 32) {
        // ---- issue all global loads for this 32-key chunk up front ----
        const float* kpt = KpL + (size_t)kb * Dk;
        float kf0[8], kf1[8];
#pragma unroll
        for (int i = 0; i < 8; ++i) { kf0[i] = kpt[i]; kf1[i] = kpt[16 * Dk + i]; }

        const float* bpt = BpL + kb;
        float bv0[8], bv1[8];
#pragma unroll
        for (int i = 0; i < 8; ++i) { bv0[i] = bpt[i]; bv1[i] = bpt[16 + i]; }

        const uint2 mw0 = *(const uint2*)(MpL + kb);      // 8 mask bytes, tile0
        const uint2 mw1 = *(const uint2*)(MpL + kb + 16); // 8 mask bytes, tile1

        // ---- K tiles as A-operands (16x32 f16, kred=d, upper half zeroed) ----
        v16h a_k0, a_k1;
#pragma unroll
        for (int i = 0; i < 8; ++i) { a_k0[i] = (_Float16)kf0[i]; a_k1[i] = (_Float16)kf1[i]; }
#pragma unroll
        for (int i = 8; i < 16; ++i) { a_k0[i] = (_Float16)0.0f; a_k1[i] = (_Float16)0.0f; }

        // ---- transposed score tiles: S^T = K_tile @ Q^T ----
        v8f c0 = {}, c1 = {};
        c0 = __builtin_amdgcn_wmma_f32_16x16x32_f16(
                 false, a_k0, false, b_q, (short)0, c0, false, false);
        c1 = __builtin_amdgcn_wmma_f32_16x16x32_f16(
                 false, a_k1, false, b_q, (short)0, c1, false, false);
        // c_t[r] = score(q = qrow, key = kb + t*16 + r + 8*half)  -> one row per lane

        // ---- bias + mask (all in-lane, contiguous) ----
        float s0[8], s1[8];
#pragma unroll
        for (int r = 0; r < 8; ++r) {
            const unsigned int w0 = (r & 4) ? mw0.y : mw0.x;
            const unsigned int w1 = (r & 4) ? mw1.y : mw1.x;
            const unsigned int m0 = (w0 >> ((r & 3) * 8)) & 0xffu;
            const unsigned int m1 = (w1 >> ((r & 3) * 8)) & 0xffu;
            s0[r] = m0 ? -1e9f : c0[r] + bv0[r];
            s1[r] = m1 ? -1e9f : c1[r] + bv1[r];
        }

        // ---- online softmax: in-lane over 16 values + one cross-half merge ----
        float mx = fmaxf(s0[0], s1[0]);
#pragma unroll
        for (int r = 1; r < 8; ++r) mx = fmaxf(mx, fmaxf(s0[r], s1[r]));
        mx = fmaxf(mx, __shfl_xor(mx, 16, 32));
        const float mn = fmaxf(m, mx);
        const float sc = __expf(m - mn);
        m = mn;

        float p0[8], p1[8], ps = 0.0f;
#pragma unroll
        for (int r = 0; r < 8; ++r) {
            p0[r] = __expf(s0[r] - mn);
            p1[r] = __expf(s1[r] - mn);
            ps += p0[r] + p1[r];
        }
        ps += __shfl_xor(ps, 16, 32);
        l = l * sc + ps;

        // ---- rescale accumulator: row (r+8h) scale lives in lane (r+8h) ----
#pragma unroll
        for (int r = 0; r < 8; ++r)
            acc[r] *= __shfl(sc, half * 8 + r, 32);

        // ---- P is ALREADY in A-operand layout: a_p[r]=p0[r], a_p[8+r]=p1[r] ----
        v16h a_p;
#pragma unroll
        for (int r = 0; r < 8; ++r) {
            a_p[r]     = (_Float16)p0[r];
            a_p[8 + r] = (_Float16)p1[r];
        }

        // ---- V as B-operand: lane = column d=row, kred = keys kb+8h..+7, kb+16+8h..+7 ----
        v16h b_v;
        {
            const _Float16* vrow = &Vt[row * VROW + kb + half * 8];
#pragma unroll
            for (int i = 0; i < 16; ++i) b_v[i] = vrow[i + (i & 8)];
        }
        acc = __builtin_amdgcn_wmma_f32_16x16x32_f16(
                  false, a_p, false, b_v, (short)0, acc, false, false);
    }

    // ---- normalize and store: acc[r] = context(q = q0+r+8h, d = row) ----
    const float rinv = 1.0f / l;
#pragma unroll
    for (int r = 0; r < 8; ++r) {
        const float o = acc[r] * __shfl(rinv, half * 8 + r, 32);
        Op[(size_t)(q0 + r + half * 8) * Dk + row] = o;
    }
}

extern "C" void kernel_launch(void* const* d_in, const int* in_sizes, int n_in,
                              void* d_out, int out_size, void* d_ws, size_t ws_size,
                              hipStream_t stream) {
    const float*         Q    = (const float*)d_in[0];
    const float*         K    = (const float*)d_in[1];
    const float*         V    = (const float*)d_in[2];
    const unsigned char* mask = (const unsigned char*)d_in[3]; // jnp.bool_ -> 1 byte
    const float*         bias = (const float*)d_in[4];
    float*               out  = (float*)d_out;

    dim3 grid(Bn * Hn * (Sn / QT)); // 1024 workgroups
    dim3 block(WAVES * 32);         // 256 threads = 8 waves
    hipLaunchKernelGGL(sdpa_flash_kernel, grid, block, 0, stream,
                       Q, K, V, mask, bias, out);
}
